// fmamba_mamba_head_18167711662218
// MI455X (gfx1250) — compile-verified, matrix operand order checked by reading, not run on hardware
//
#include <hip/hip_runtime.h>
#include <hip/hip_bf16.h>
#include <math.h>

// ---------------- constants (from reference) ----------------
#define B_  8
#define L_  2048
#define D_  1024
#define ED_ 2048
#define N_  64
#define R_  64
#define NL_ 3
#define NC_ 2
#define M_  (B_ * L_)          // 16384 rows for all GEMMs

typedef __bf16 bf16_t;
typedef __bf16 v16bf __attribute__((ext_vector_type(16)));
typedef __bf16 v8bf  __attribute__((ext_vector_type(8)));
typedef float  v8f   __attribute__((ext_vector_type(8)));

// ---------------- elementwise helpers ----------------
__global__ __launch_bounds__(256) void k_f32_to_bf16(const float* __restrict__ in,
                                                     bf16_t* __restrict__ out, size_t n) {
  size_t i = (size_t)blockIdx.x * blockDim.x + threadIdx.x;
  if (i < n) out[i] = (bf16_t)in[i];
}

__global__ __launch_bounds__(256) void k_copy_f32(const float* __restrict__ in,
                                                  float* __restrict__ out, size_t n) {
  size_t i = (size_t)blockIdx.x * blockDim.x + threadIdx.x;
  if (i < n) out[i] = in[i];
}

__global__ __launch_bounds__(256) void k_neg_exp(const float* __restrict__ in,
                                                 float* __restrict__ out, size_t n) {
  size_t i = (size_t)blockIdx.x * blockDim.x + threadIdx.x;
  if (i < n) out[i] = -__expf(in[i]);
}

// ---------------- bf16 WMMA GEMM: C(M,N) = A(M,K) @ B(N,K)^T ----------------
// 256 threads = 8 waves in a 4(M) x 2(N) grid; each wave owns a 32x32 output
// (2x2 WMMA tiles -> 4 f32 accumulators), so per 32-wide K step it loads
// 2 A-fragments + 2 B-fragments (8x global_load_b128) and issues 4 WMMAs,
// with explicit double buffering so the s_wait before the WMMAs covers loads
// issued a full K-step earlier.
// Fragment layout per the CDNA5 16-bit 16x32 A/B VGPR spec:
//   lanes 0-15 : row = lane,    K = {0..7, 16..23}
//   lanes 16-31: row = lane-16, K = {8..15, 24..31}
__device__ inline v16bf ldfrag(const bf16_t* p) {
  v8bf lo = *(const v8bf*)(p);
  v8bf hi = *(const v8bf*)(p + 16);
  v16bf f;
#pragma unroll
  for (int i = 0; i < 8; ++i) { f[i] = lo[i]; f[i + 8] = hi[i]; }
  return f;
}

__device__ inline v8f wmma_bf16(const v16bf& a, const v16bf& b, const v8f& c) {
  return __builtin_amdgcn_wmma_f32_16x16x32_bf16(false, a, false, b,
                                                 (short)0, c, false, false);
}

// EPI: 0 = plain store, 1 = softplus(v + bias[col]) (dt-projection epilogue)
template <int EPI>
__global__ __launch_bounds__(256)
void k_gemm_bf16_wmma(const bf16_t* __restrict__ A, const bf16_t* __restrict__ Bw,
                      float* __restrict__ C, const float* __restrict__ bias,
                      int M, int N, int K) {
  const int lane   = threadIdx.x & 31;
  const int wid    = threadIdx.x >> 5;       // 0..7
  const int wm     = wid & 3;                // wave row (0..3)
  const int wn     = wid >> 2;               // wave col (0..1)
  const int m_base = blockIdx.y * 128 + wm * 32;
  const int n_base = blockIdx.x * 64  + wn * 32;
  const int half   = lane >> 4;              // K-half select
  const int r      = lane & 15;

  const bf16_t* a0p = A  + (size_t)(m_base + r)      * K + half * 8;
  const bf16_t* a1p = A  + (size_t)(m_base + 16 + r) * K + half * 8;
  const bf16_t* b0p = Bw + (size_t)(n_base + r)      * K + half * 8;
  const bf16_t* b1p = Bw + (size_t)(n_base + 16 + r) * K + half * 8;

  v8f c00 = {}, c01 = {}, c10 = {}, c11 = {};
  v16bf ca0, ca1, cb0, cb1;     // current K-step fragments
  v16bf na0, na1, nb0, nb1;     // next K-step fragments

  ca0 = ldfrag(a0p); ca1 = ldfrag(a1p); cb0 = ldfrag(b0p); cb1 = ldfrag(b1p);

  int kk = 0;
  for (; kk + 64 < K; kk += 64) {            // K is a multiple of 64 for all calls
    na0 = ldfrag(a0p + kk + 32); na1 = ldfrag(a1p + kk + 32);
    nb0 = ldfrag(b0p + kk + 32); nb1 = ldfrag(b1p + kk + 32);
    if (kk + 256 < K) {                      // global_prefetch_b8, 4 streams
      __builtin_prefetch(a0p + kk + 256, 0, 1);
      __builtin_prefetch(a1p + kk + 256, 0, 1);
      __builtin_prefetch(b0p + kk + 256, 0, 1);
      __builtin_prefetch(b1p + kk + 256, 0, 1);
    }
    c00 = wmma_bf16(ca0, cb0, c00);
    c01 = wmma_bf16(ca0, cb1, c01);
    c10 = wmma_bf16(ca1, cb0, c10);
    c11 = wmma_bf16(ca1, cb1, c11);

    ca0 = ldfrag(a0p + kk + 64); ca1 = ldfrag(a1p + kk + 64);
    cb0 = ldfrag(b0p + kk + 64); cb1 = ldfrag(b1p + kk + 64);
    c00 = wmma_bf16(na0, nb0, c00);
    c01 = wmma_bf16(na0, nb1, c01);
    c10 = wmma_bf16(na1, nb0, c10);
    c11 = wmma_bf16(na1, nb1, c11);
  }
  // tail: kk == K - 64
  na0 = ldfrag(a0p + kk + 32); na1 = ldfrag(a1p + kk + 32);
  nb0 = ldfrag(b0p + kk + 32); nb1 = ldfrag(b1p + kk + 32);
  c00 = wmma_bf16(ca0, cb0, c00);
  c01 = wmma_bf16(ca0, cb1, c01);
  c10 = wmma_bf16(ca1, cb0, c10);
  c11 = wmma_bf16(ca1, cb1, c11);
  c00 = wmma_bf16(na0, nb0, c00);
  c01 = wmma_bf16(na0, nb1, c01);
  c10 = wmma_bf16(na1, nb0, c10);
  c11 = wmma_bf16(na1, nb1, c11);

  // C/D layout: VGPR j -> M = j (lanes 0-15) or 8+j (lanes 16-31); N = lane&15
  const int row_off = half * 8;
  auto store_tile = [&](v8f& acc, int mb, int nb) {
    const int col = nb + r;
#pragma unroll
    for (int j = 0; j < 8; ++j) {
      float v = acc[j];
      if (EPI == 1) {
        v += bias[col];
        v = (v > 20.0f) ? v : log1pf(__expf(v));
      }
      C[(size_t)(mb + row_off + j) * N + col] = v;
    }
  };
  store_tile(c00, m_base,      n_base);
  store_tile(c01, m_base,      n_base + 16);
  store_tile(c10, m_base + 16, n_base);
  store_tile(c11, m_base + 16, n_base + 16);
}

// ---------------- depthwise causal conv (K=4) + SiLU; silu(z) in place ----------------
__global__ __launch_bounds__(256)
void k_conv_silu(const float* __restrict__ cw, const float* __restrict__ cb,
                 float* __restrict__ xz, float* __restrict__ xp,
                 bf16_t* __restrict__ xpb) {
  size_t idx = (size_t)blockIdx.x * blockDim.x + threadIdx.x;   // over B*L*ED
  if (idx >= (size_t)M_ * ED_) return;
  const int    e  = (int)(idx % ED_);
  const size_t bt = idx / ED_;            // b*L + t
  const int    t  = (int)(bt % L_);
  const size_t b  = bt / L_;

  const size_t rs   = 2 * ED_;
  const float* base = xz + (size_t)b * L_ * rs + e;

  float acc = cb[e];
#pragma unroll
  for (int k = 0; k < 4; ++k) {
    int tt = t + k - 3;
    if (tt >= 0) acc += cw[e * 4 + k] * base[(size_t)tt * rs];
  }
  float s = acc / (1.0f + __expf(-acc));   // silu
  xp[bt * ED_ + e]  = s;
  xpb[bt * ED_ + e] = (bf16_t)s;

  float z = xz[bt * rs + ED_ + e];
  xz[bt * rs + ED_ + e] = z / (1.0f + __expf(-z));
}

// ---------------- dt slice -> bf16 ----------------
__global__ __launch_bounds__(256)
void k_slice_dt(const float* __restrict__ xdbl, bf16_t* __restrict__ dtb) {
  size_t idx = (size_t)blockIdx.x * blockDim.x + threadIdx.x;   // over M*R
  if (idx >= (size_t)M_ * R_) return;
  size_t row = idx / R_;
  int    n   = (int)(idx % R_);
  dtb[idx] = (bf16_t)xdbl[row * (R_ + 2 * N_) + n];
}

// ---------------- selective scan ----------------
// one thread per (b,e); 64 states + 64 A values in registers. B_t/C_t rows are
// wave-uniform (lanes share b,t) -> hardware-broadcast loads.
__global__ __launch_bounds__(64)
void k_scan(const float* __restrict__ delta, const float* __restrict__ xdbl,
            const float* __restrict__ xp, const float* __restrict__ xz,
            const float* __restrict__ Aneg, const float* __restrict__ Dp,
            bf16_t* __restrict__ yb) {
  int tid = blockIdx.x * blockDim.x + threadIdx.x;   // B*ED threads
  int e = tid % ED_;
  int b = tid / ED_;

  float An[N_], h[N_];
#pragma unroll
  for (int n = 0; n < N_; ++n) { An[n] = Aneg[e * N_ + n]; h[n] = 0.0f; }
  const float dpe = Dp[e];

  for (int t = 0; t < L_; ++t) {
    size_t bt = (size_t)b * L_ + t;
    float d  = delta[bt * ED_ + e];
    float xt = xp[bt * ED_ + e];
    float dx = d * xt;
    const float* Brow = xdbl + bt * (R_ + 2 * N_) + R_;
    const float* Crow = Brow + N_;
    if (t + 1 < L_) __builtin_prefetch(Brow + (R_ + 2 * N_), 0, 1);
    float y = 0.0f;
#pragma unroll
    for (int n = 0; n < N_; ++n) {
      float hb = __expf(d * An[n]) * h[n] + dx * Brow[n];
      h[n] = hb;
      y += hb * Crow[n];
    }
    y += xt * dpe;
    float sz = xz[bt * (2 * ED_) + ED_ + e];   // silu(z), precomputed in place
    yb[bt * ED_ + e] = (bf16_t)(y * sz);
  }
}

// ---------------- rmsnorm + residual (x updated in place) ----------------
__global__ __launch_bounds__(256)
void k_rmsnorm_residual(const float* __restrict__ zin, const float* __restrict__ w,
                        float* __restrict__ x) {
  const int row = blockIdx.x;                 // over M_ rows of D_
  const float* zr = zin + (size_t)row * D_;
  float ss = 0.0f;
  for (int i = threadIdx.x; i < D_; i += 256) { float v = zr[i]; ss += v * v; }
#pragma unroll
  for (int off = 16; off; off >>= 1) ss += __shfl_down(ss, off, 32);
  __shared__ float red[8];
  if ((threadIdx.x & 31) == 0) red[threadIdx.x >> 5] = ss;
  __syncthreads();
  float tot = 0.0f;
#pragma unroll
  for (int i = 0; i < 8; ++i) tot += red[i];
  float rs = rsqrtf(tot / (float)D_ + 1e-5f);
  for (int i = threadIdx.x; i < D_; i += 256) {
    size_t o = (size_t)row * D_ + i;
    x[o] = zr[i] * rs * w[i] + x[o];
  }
}

// ---------------- head: out(b,c) = x[b, L-1, :] . W_head[c, :] + b_head[c] ----------------
__global__ __launch_bounds__(256)
void k_head(const float* __restrict__ x, const float* __restrict__ Wh,
            const float* __restrict__ bh, float* __restrict__ out) {
  int b = blockIdx.x / NC_;
  int c = blockIdx.x % NC_;
  const float* xr = x + ((size_t)b * L_ + (L_ - 1)) * D_;
  const float* wr = Wh + (size_t)c * D_;
  float s = 0.0f;
  for (int i = threadIdx.x; i < D_; i += 256) s += xr[i] * wr[i];
#pragma unroll
  for (int off = 16; off; off >>= 1) s += __shfl_down(s, off, 32);
  __shared__ float red[8];
  if ((threadIdx.x & 31) == 0) red[threadIdx.x >> 5] = s;
  __syncthreads();
  if (threadIdx.x == 0) {
    float t = 0.0f;
#pragma unroll
    for (int i = 0; i < 8; ++i) t += red[i];
    out[b * NC_ + c] = t + bh[c];
  }
}

// ---------------- host ----------------
extern "C" void kernel_launch(void* const* d_in, const int* in_sizes, int n_in,
                              void* d_out, int out_size, void* d_ws, size_t ws_size,
                              hipStream_t stream) {
  (void)in_sizes; (void)n_in; (void)out_size; (void)ws_size;
  const float* x_in   = (const float*)d_in[0];
  const float* W_in   = (const float*)d_in[1];   // (NL, 2*ED, D)
  const float* conv_w = (const float*)d_in[2];   // (NL, ED, 1, 4)
  const float* conv_b = (const float*)d_in[3];   // (NL, ED)
  const float* W_x    = (const float*)d_in[4];   // (NL, R+2N, ED)
  const float* W_dt   = (const float*)d_in[5];   // (NL, ED, R)
  const float* b_dt   = (const float*)d_in[6];   // (NL, ED)
  const float* A_log  = (const float*)d_in[7];   // (NL, ED, N)
  const float* D_p    = (const float*)d_in[8];   // (NL, ED)
  const float* W_out  = (const float*)d_in[9];   // (NL, D, ED)
  const float* norm_w = (const float*)d_in[10];  // (NL, D)
  const float* W_head = (const float*)d_in[11];  // (NC, D)
  const float* b_head = (const float*)d_in[12];  // (NC,)
  float* out = (float*)d_out;

  // ---- bump allocator on workspace ----
  char* base = (char*)d_ws;
  size_t off = 0;
  auto alloc = [&](size_t bytes) -> void* {
    size_t a = (off + 255) & ~(size_t)255;
    off = a + bytes;
    return (void*)(base + a);
  };

  bf16_t* wbin  = (bf16_t*)alloc((size_t)NL_ * 2 * ED_ * D_  * sizeof(bf16_t));
  bf16_t* wbx   = (bf16_t*)alloc((size_t)NL_ * (R_ + 2 * N_) * ED_ * sizeof(bf16_t));
  bf16_t* wbdt  = (bf16_t*)alloc((size_t)NL_ * ED_ * R_      * sizeof(bf16_t));
  bf16_t* wbout = (bf16_t*)alloc((size_t)NL_ * D_  * ED_     * sizeof(bf16_t));
  float*  aneg  = (float*) alloc((size_t)NL_ * ED_ * N_      * sizeof(float));
  float*  xcur  = (float*) alloc((size_t)M_ * D_             * sizeof(float));
  bf16_t* actbf = (bf16_t*)alloc((size_t)M_ * ED_            * sizeof(bf16_t)); // xb / xp_bf / y_bf
  float*  xz    = (float*) alloc((size_t)M_ * 2 * ED_        * sizeof(float));
  float*  xp    = (float*) alloc((size_t)M_ * ED_            * sizeof(float));
  float*  xdbl  = (float*) alloc((size_t)M_ * (R_ + 2 * N_)  * sizeof(float));
  bf16_t* dtb   = (bf16_t*)alloc((size_t)M_ * R_             * sizeof(bf16_t));
  float*  dbuf  = (float*) alloc((size_t)M_ * ED_            * sizeof(float));  // delta, then out-proj result

  auto nblk = [](size_t n) { return (unsigned)((n + 255) / 256); };
  auto cvt = [&](const float* s, bf16_t* d, size_t n) {
    k_f32_to_bf16<<<nblk(n), 256, 0, stream>>>(s, d, n);
  };
  auto gemm = [&](const bf16_t* Aa, const bf16_t* Bb, float* Cc, int M, int N, int K) {
    dim3 g((unsigned)(N / 64), (unsigned)(M / 128));
    k_gemm_bf16_wmma<0><<<g, 256, 0, stream>>>(Aa, Bb, Cc, nullptr, M, N, K);
  };

  // ---- one-time (per launch) weight conversion + A = -exp(A_log) ----
  cvt(W_in,  wbin,  (size_t)NL_ * 2 * ED_ * D_);
  cvt(W_x,   wbx,   (size_t)NL_ * (R_ + 2 * N_) * ED_);
  cvt(W_dt,  wbdt,  (size_t)NL_ * ED_ * R_);
  cvt(W_out, wbout, (size_t)NL_ * D_ * ED_);
  k_neg_exp<<<nblk((size_t)NL_ * ED_ * N_), 256, 0, stream>>>(A_log, aneg, (size_t)NL_ * ED_ * N_);
  k_copy_f32<<<nblk((size_t)M_ * D_), 256, 0, stream>>>(x_in, xcur, (size_t)M_ * D_);

  for (int i = 0; i < NL_; ++i) {
    // in-projection: xz = x @ W_in^T   (16384 x 4096 x 1024)
    cvt(xcur, actbf, (size_t)M_ * D_);
    gemm(actbf, wbin + (size_t)i * 2 * ED_ * D_, xz, M_, 2 * ED_, D_);

    // depthwise causal conv + silu; silu(z) in place
    k_conv_silu<<<nblk((size_t)M_ * ED_), 256, 0, stream>>>(
        conv_w + (size_t)i * ED_ * 4, conv_b + (size_t)i * ED_, xz, xp, actbf);

    // x-projection: xdbl = xp @ W_x^T  (16384 x 192 x 2048)
    gemm(actbf, wbx + (size_t)i * (R_ + 2 * N_) * ED_, xdbl, M_, R_ + 2 * N_, ED_);

    // dt-projection with fused bias+softplus epilogue  (16384 x 2048 x 64)
    k_slice_dt<<<nblk((size_t)M_ * R_), 256, 0, stream>>>(xdbl, dtb);
    {
      dim3 g(ED_ / 64, M_ / 128);
      k_gemm_bf16_wmma<1><<<g, 256, 0, stream>>>(
          dtb, wbdt + (size_t)i * ED_ * R_, dbuf, b_dt + (size_t)i * ED_, M_, ED_, R_);
    }

    // selective scan -> y * silu(z) as bf16 (into actbf, xp_bf no longer needed)
    k_scan<<<(B_ * ED_) / 64, 64, 0, stream>>>(
        dbuf, xdbl, xp, xz, aneg + (size_t)i * ED_ * N_, D_p + (size_t)i * ED_, actbf);

    // out-projection: z = y @ W_out^T  (16384 x 1024 x 2048) -> dbuf (delta dead)
    gemm(actbf, wbout + (size_t)i * D_ * ED_, dbuf, M_, D_, ED_);

    // x = rmsnorm(z) + x
    k_rmsnorm_residual<<<M_, 256, 0, stream>>>(dbuf, norm_w + (size_t)i * D_, xcur);
  }

  // head on last token
  k_head<<<B_ * NC_, 256, 0, stream>>>(xcur, W_head, b_head, out);
}
